// MambaBlock_46909632807182
// MI455X (gfx1250) — matrix operand order repaired and checked by reading, not compiled, not run on hardware
//
#include <hip/hip_runtime.h>
#include <hip/hip_bf16.h>

// ---------------- problem constants (from reference) ----------------
#define B_   2
#define L_   2048
#define DM_  2048
#define DI_  4096
#define DS_  16
#define DTR_ 128
#define DC_  4
#define E_   (2*DI_)        // 8192
#define M_   (B_*L_)        // 4096 rows for all GEMMs
#define XDW_ (DTR_ + 2*DS_) // 160
#define PCH_ 16             // scan chunks
#define CL_  (L_/PCH_)      // 128 steps per chunk

typedef __attribute__((ext_vector_type(16))) __bf16 v16bf;
typedef __attribute__((ext_vector_type(8)))  float  v8f;

union FragU { uint4 q[2]; v16bf v; };

__device__ __forceinline__ unsigned short f2bf(float f) {
    unsigned int u = __float_as_uint(f);
    unsigned int r = u + 0x7fffu + ((u >> 16) & 1u);   // RNE
    return (unsigned short)(r >> 16);
}
__device__ __forceinline__ float bf2f(unsigned short h) {
    return __uint_as_float(((unsigned int)h) << 16);
}
__device__ __forceinline__ float softplusf(float x) {
    return (x > 20.f) ? x : log1pf(__expf(x));
}
__device__ __forceinline__ float sigmoidf_(float x) {
    return 1.f / (1.f + __expf(-x));
}

// A-matrix 16x32 bf16 frag: lane half h: VGPR0-3 = K[h*8..h*8+7], VGPR4-7 = K[h*8+16..h*8+23]
__device__ __forceinline__ v16bf load_fragA(const unsigned short* p) {
    FragU f;
    f.q[0] = *(const uint4*)(p);
    f.q[1] = *(const uint4*)(p + 16);
    return f.v;
}
// B-matrix 32x16 bf16 frag: lane half h: VGPR0-3 = K[h*8..h*8+7], VGPR4-7 = K[h*8+16..h*8+23]
__device__ __forceinline__ v16bf load_fragB(const unsigned short* p) {
    FragU f;
    f.q[0] = *(const uint4*)(p);
    f.q[1] = *(const uint4*)(p + 16);
    return f.v;
}

// ---------------- fp32 -> bf16 converts (4-wide) ----------------
__global__ void cvt_f32_bf16(const float* __restrict__ s,
                             unsigned short* __restrict__ d, size_t n4) {
    size_t i = (size_t)blockIdx.x * 256 + threadIdx.x;
    size_t stride = (size_t)gridDim.x * 256;
    for (; i < n4; i += stride) {
        float4 v = ((const float4*)s)[i];
        unsigned int lo = (unsigned)f2bf(v.x) | ((unsigned)f2bf(v.y) << 16);
        unsigned int hi = (unsigned)f2bf(v.z) | ((unsigned)f2bf(v.w) << 16);
        ((uint2*)d)[i] = make_uint2(lo, hi);
    }
}

// first `cols` columns (cols%4==0) of [rows, srcw] f32 -> bf16 [rows, cols]
__global__ void cvt_slice_bf16(const float* __restrict__ s,
                               unsigned short* __restrict__ d,
                               int rows, int srcw, int cols) {
    size_t n4 = (size_t)rows * (cols / 4);
    size_t i = (size_t)blockIdx.x * 256 + threadIdx.x;
    size_t stride = (size_t)gridDim.x * 256;
    for (; i < n4; i += stride) {
        int c4 = (int)(i % (cols / 4));
        int r  = (int)(i / (cols / 4));
        float4 v = *(const float4*)(s + (size_t)r * srcw + c4 * 4);
        unsigned int lo = (unsigned)f2bf(v.x) | ((unsigned)f2bf(v.y) << 16);
        unsigned int hi = (unsigned)f2bf(v.z) | ((unsigned)f2bf(v.w) << 16);
        *(uint2*)(d + (size_t)r * cols + c4 * 4) = make_uint2(lo, hi);
    }
}

// ---------------- generic bf16 NT WMMA GEMM ----------------
// C[M,N](f32) = A[M,K](bf16 rm) * B[N,K](bf16 rm)^T ; op==1: softplus(acc + bias[col])
#define BM 128
#define BN 128
__global__ __launch_bounds__(256)
void gemm_bf16_nt(const unsigned short* __restrict__ A,
                  const unsigned short* __restrict__ Bm,
                  float* __restrict__ C,
                  int M, int N, int K,
                  const float* __restrict__ bias, int op) {
    const int tid  = threadIdx.x;
    const int wave = tid >> 5;
    const int lane = tid & 31;
    const int half = lane >> 4;
    const int r16  = lane & 15;

    const int bm = blockIdx.y * BM;
    const int bn = blockIdx.x * BN;
    const int wm = wave & 3;          // 4 waves along M
    const int wn = wave >> 2;         // 2 waves along N

    v8f acc[2][4] = {};
    const int kha = half * 8;
    const int khb = half * 8;

    for (int k = 0; k < K; k += 32) {
        v16bf af[2], bf[4];
#pragma unroll
        for (int i = 0; i < 2; i++) {
            int row = bm + wm * 32 + i * 16 + r16;
            af[i] = load_fragA(A + (size_t)row * K + k + kha);
        }
#pragma unroll
        for (int j = 0; j < 4; j++) {
            int row = bn + wn * 64 + j * 16 + r16;
            if (row >= N) row = N - 1;
            bf[j] = load_fragB(Bm + (size_t)row * K + k + khb);
        }
#pragma unroll
        for (int i = 0; i < 2; i++)
#pragma unroll
            for (int j = 0; j < 4; j++)
                acc[i][j] = __builtin_amdgcn_wmma_f32_16x16x32_bf16(
                    false, af[i], false, bf[j], (short)0, acc[i][j], false, false);
    }

#pragma unroll
    for (int i = 0; i < 2; i++) {
#pragma unroll
        for (int j = 0; j < 4; j++) {
            int col = bn + wn * 64 + j * 16 + r16;
            if (col < N) {
#pragma unroll
                for (int v = 0; v < 8; v++) {
                    int row = bm + wm * 32 + i * 16 + half * 8 + v;
                    float x = acc[i][j][v];
                    if (op == 1) x = softplusf(x + bias[col]);
                    C[(size_t)row * N + col] = x;
                }
            }
        }
    }
}

// ---------------- causal depthwise conv (DC=4) + SiLU -> bf16 ----------------
__global__ void conv_silu(const float* __restrict__ xz,
                          const float* __restrict__ cw,
                          const float* __restrict__ cb,
                          unsigned short* __restrict__ xbf) {
    size_t n = (size_t)B_ * L_ * DI_;
    size_t i = (size_t)blockIdx.x * 256 + threadIdx.x;
    if (i >= n) return;
    int d = (int)(i % DI_);
    int l = (int)((i / DI_) % L_);
    int b = (int)(i / ((size_t)DI_ * L_));
    size_t rowb = (size_t)b * L_;
    float acc = cb[d];
#pragma unroll
    for (int j = 0; j < DC_; j++) {
        int ll = l - (DC_ - 1) + j;
        if (ll >= 0) acc += cw[d * DC_ + j] * xz[(rowb + ll) * E_ + d];
    }
    float s = acc * sigmoidf_(acc);
    xbf[i] = f2bf(s);
}

// ---------------- chunked selective scan ----------------
// Pass A: per (b, chunk, d): local scan from 0 -> S_loc[16], sum_dt
__global__ __launch_bounds__(256)
void scan_partial(const float* __restrict__ delta,
                  const unsigned short* __restrict__ u_bf,
                  const float* __restrict__ xdbl,
                  const float* __restrict__ negA,
                  float* __restrict__ Sloc,
                  float* __restrict__ sumdt) {
    __shared__ __align__(16) float sB[CL_ * DS_];     // 8 KB
    const int bid = blockIdx.x;
    const int b  = bid / (PCH_ * 16);
    const int c  = (bid / 16) % PCH_;
    const int dg = bid % 16;
    const int d  = (dg << 8) + threadIdx.x;
    const size_t rowb = (size_t)b * L_ + (size_t)c * CL_;

    // stage B rows: CL_ rows x 16 floats (64B per row, 16B-aligned pieces)
    for (int idx = threadIdx.x; idx < CL_ * 4; idx += 256) {
        int row = idx >> 2, part = idx & 3;
        uint4 v = *(const uint4*)(xdbl + (rowb + row) * XDW_ + DTR_ + part * 4);
        *(uint4*)&sB[row * DS_ + part * 4] = v;
    }
    __syncthreads();

    float a[DS_], s[DS_];
#pragma unroll
    for (int n = 0; n < DS_; n++) { a[n] = negA[(size_t)d * DS_ + n]; s[n] = 0.f; }
    float sdt = 0.f;

    for (int li = 0; li < CL_; li++) {
        size_t row = rowb + li;
        float dt = delta[row * DI_ + d];
        float u  = bf2f(u_bf[row * DI_ + d]);
        float du = dt * u;
        sdt += dt;
#pragma unroll
        for (int n = 0; n < DS_; n++)
            s[n] = s[n] * __expf(dt * a[n]) + du * sB[li * DS_ + n];
    }
    size_t o = ((size_t)b * DI_ + d) * PCH_ + c;
#pragma unroll
    for (int n = 0; n < DS_; n++) Sloc[o * DS_ + n] = s[n];
    sumdt[o] = sdt;
}

// Pass B: per (b, d): combine chunks; write per-chunk initial state + last_state
__global__ __launch_bounds__(256)
void scan_combine(const float* __restrict__ Sloc,
                  const float* __restrict__ sumdt,
                  const float* __restrict__ negA,
                  float* __restrict__ s_in,
                  float* __restrict__ last_state) {
    int t = blockIdx.x * 256 + threadIdx.x;      // 0..B_*DI_-1
    int b = t / DI_, d = t % DI_;
    float a[DS_], s[DS_];
#pragma unroll
    for (int n = 0; n < DS_; n++) { a[n] = negA[(size_t)d * DS_ + n]; s[n] = 0.f; }
    size_t base = ((size_t)b * DI_ + d) * PCH_;
    for (int c = 0; c < PCH_; c++) {
#pragma unroll
        for (int n = 0; n < DS_; n++) s_in[(base + c) * DS_ + n] = s[n];
        float sdt = sumdt[base + c];
#pragma unroll
        for (int n = 0; n < DS_; n++)
            s[n] = s[n] * __expf(a[n] * sdt) + Sloc[(base + c) * DS_ + n];
    }
#pragma unroll
    for (int n = 0; n < DS_; n++)
        last_state[((size_t)b * DI_ + d) * DS_ + n] = s[n];
}

// Pass C: exact scan per chunk from corrected initial state; fused skip+gate -> gy bf16
__global__ __launch_bounds__(256)
void scan_final(const float* __restrict__ delta,
                const unsigned short* __restrict__ u_bf,
                const float* __restrict__ xdbl,
                const float* __restrict__ xz,
                const float* __restrict__ negA,
                const float* __restrict__ Dp,
                const float* __restrict__ s_in,
                unsigned short* __restrict__ gy) {
    __shared__ __align__(16) float sBC[CL_ * 32];    // 16 KB: [li][0:16)=B, [16:32)=C
    const int bid = blockIdx.x;
    const int b  = bid / (PCH_ * 16);
    const int c  = (bid / 16) % PCH_;
    const int dg = bid % 16;
    const int d  = (dg << 8) + threadIdx.x;
    const size_t rowb = (size_t)b * L_ + (size_t)c * CL_;

    // async stage B|C rows: CL_ rows x 32 floats (128B/row, 16B-aligned eighths)
    unsigned ldsbase = (unsigned)(size_t)sBC;
    for (int k = 0; k < 4; k++) {
        int idx  = threadIdx.x + k * 256;            // 0..1023
        int row  = idx >> 3, part = idx & 7;
        const float* gp = xdbl + (rowb + row) * XDW_ + DTR_ + part * 4;
        unsigned lo = ldsbase + (unsigned)(row * 32 + part * 4) * 4u;
        asm volatile("global_load_async_to_lds_b128 %0, %1, off"
                     :: "v"(lo), "v"(gp) : "memory");
    }
    asm volatile("s_wait_asynccnt 0x0" ::: "memory");
    __syncthreads();

    float a[DS_], s[DS_];
    size_t sbase = (((size_t)b * DI_ + d) * PCH_ + c) * DS_;
#pragma unroll
    for (int n = 0; n < DS_; n++) {
        a[n] = negA[(size_t)d * DS_ + n];
        s[n] = s_in[sbase + n];
    }
    const float Dd = Dp[d];

    for (int li = 0; li < CL_; li++) {
        size_t row = rowb + li;
        float dt = delta[row * DI_ + d];
        float u  = bf2f(u_bf[row * DI_ + d]);
        float du = dt * u;
        float y  = 0.f;
#pragma unroll
        for (int n = 0; n < DS_; n++) {
            s[n] = s[n] * __expf(dt * a[n]) + du * sBC[li * 32 + n];
            y += s[n] * sBC[li * 32 + 16 + n];
        }
        float z = xz[row * E_ + DI_ + d];
        float out = (y + Dd * u) * (z * sigmoidf_(z));
        gy[row * DI_ + d] = f2bf(out);
    }
}

// ---------------- host orchestration ----------------
static inline size_t alignup(size_t x) { return (x + 255) & ~(size_t)255; }

extern "C" void kernel_launch(void* const* d_in, const int* in_sizes, int n_in,
                              void* d_out, int out_size, void* d_ws, size_t ws_size,
                              hipStream_t stream) {
    const float* hs    = (const float*)d_in[0];
    const float* inw   = (const float*)d_in[1];
    const float* convw = (const float*)d_in[2];
    const float* convb = (const float*)d_in[3];
    const float* xpw   = (const float*)d_in[4];
    const float* dtw   = (const float*)d_in[5];
    const float* dtb   = (const float*)d_in[6];
    const float* outw  = (const float*)d_in[7];
    const float* negA  = (const float*)d_in[8];
    const float* Dp    = (const float*)d_in[9];

    float* out_main  = (float*)d_out;
    float* out_state = (float*)d_out + (size_t)B_ * L_ * DM_;

    char* p = (char*)d_ws;  size_t off = 0;
    auto take = [&](size_t bytes) { char* r = p + off; off = alignup(off + bytes); return r; };
    unsigned short* hs_bf   = (unsigned short*)take((size_t)M_ * DM_ * 2);
    unsigned short* inw_bf  = (unsigned short*)take((size_t)E_ * DM_ * 2);
    unsigned short* xpw_bf  = (unsigned short*)take((size_t)XDW_ * DI_ * 2);
    unsigned short* dtw_bf  = (unsigned short*)take((size_t)DI_ * DTR_ * 2);
    unsigned short* outw_bf = (unsigned short*)take((size_t)DM_ * DI_ * 2);
    float*          xz      = (float*)take((size_t)M_ * E_ * 4);
    unsigned short* x_bf    = (unsigned short*)take((size_t)M_ * DI_ * 2);
    float*          xdbl    = (float*)take((size_t)M_ * XDW_ * 4);
    unsigned short* dt_bf   = (unsigned short*)take((size_t)M_ * DTR_ * 2);
    float*          deltaB  = (float*)take((size_t)M_ * DI_ * 4);
    unsigned short* gy_bf   = (unsigned short*)take((size_t)M_ * DI_ * 2);
    float*          Sloc    = (float*)take((size_t)B_ * DI_ * PCH_ * DS_ * 4);
    float*          s_in    = (float*)take((size_t)B_ * DI_ * PCH_ * DS_ * 4);
    float*          sumdt   = (float*)take((size_t)B_ * DI_ * PCH_ * 4);
    (void)ws_size; (void)n_in; (void)in_sizes; (void)out_size;

    auto cvt = [&](const float* s, unsigned short* d, size_t n) {
        size_t n4 = n / 4;
        int blocks = (int)(((n4 + 255) / 256 < 4096) ? (n4 + 255) / 256 : 4096);
        hipLaunchKernelGGL(cvt_f32_bf16, dim3(blocks), dim3(256), 0, stream, s, d, n4);
    };
    cvt(hs,   hs_bf,   (size_t)M_ * DM_);
    cvt(inw,  inw_bf,  (size_t)E_ * DM_);
    cvt(xpw,  xpw_bf,  (size_t)XDW_ * DI_);
    cvt(dtw,  dtw_bf,  (size_t)DI_ * DTR_);
    cvt(outw, outw_bf, (size_t)DM_ * DI_);

    // 1) in_proj
    hipLaunchKernelGGL(gemm_bf16_nt, dim3(E_ / BN, M_ / BM), dim3(256), 0, stream,
                       hs_bf, inw_bf, xz, M_, E_, DM_, (const float*)nullptr, 0);
    // 2) conv + SiLU
    {
        size_t n = (size_t)M_ * DI_;
        hipLaunchKernelGGL(conv_silu, dim3((unsigned)((n + 255) / 256)), dim3(256), 0, stream,
                           xz, convw, convb, x_bf);
    }
    // 3) x_proj
    hipLaunchKernelGGL(gemm_bf16_nt, dim3((XDW_ + BN - 1) / BN, M_ / BM), dim3(256), 0, stream,
                       x_bf, xpw_bf, xdbl, M_, XDW_, DI_, (const float*)nullptr, 0);
    // 4) dt slice -> bf16
    {
        size_t n4 = (size_t)M_ * (DTR_ / 4);
        int blocks = (int)(((n4 + 255) / 256 < 4096) ? (n4 + 255) / 256 : 4096);
        hipLaunchKernelGGL(cvt_slice_bf16, dim3(blocks), dim3(256), 0, stream,
                           xdbl, dt_bf, M_, XDW_, DTR_);
    }
    // 5) dt_proj + bias + softplus
    hipLaunchKernelGGL(gemm_bf16_nt, dim3(DI_ / BN, M_ / BM), dim3(256), 0, stream,
                       dt_bf, dtw_bf, deltaB, M_, DI_, DTR_, dtb, 1);

    // 6) chunked scan
    const int scan_blocks = B_ * PCH_ * (DI_ / 256);   // 512
    hipLaunchKernelGGL(scan_partial, dim3(scan_blocks), dim3(256), 0, stream,
                       deltaB, x_bf, xdbl, negA, Sloc, sumdt);
    hipLaunchKernelGGL(scan_combine, dim3((B_ * DI_) / 256), dim3(256), 0, stream,
                       Sloc, sumdt, negA, s_in, out_state);
    hipLaunchKernelGGL(scan_final, dim3(scan_blocks), dim3(256), 0, stream,
                       deltaB, x_bf, xdbl, xz, negA, Dp, s_in, gy_bf);

    // 7) out_proj
    hipLaunchKernelGGL(gemm_bf16_nt, dim3(DM_ / BN, M_ / BM), dim3(256), 0, stream,
                       gy_bf, outw_bf, out_main, M_, DM_, DI_, (const float*)nullptr, 0);
}
